// Y2DLSTMCell_35158602285339
// MI455X (gfx1250) — matrix operand order repaired and checked by reading, not compiled, not run on hardware
//
#include <hip/hip_runtime.h>
#include <hip/hip_bf16.h>

typedef __attribute__((ext_vector_type(16))) __bf16        v16bf;
typedef __attribute__((ext_vector_type(8)))  float         v8f;
typedef __attribute__((ext_vector_type(8)))  unsigned int  v8u;

#define UDIM   512
#define NCOL   2560          // 5*U
#define KDIM   1536          // 3*U
#define MTILE  32            // rows per workgroup (2 row-tiles per wave)
#define RT     2             // row-tiles per wave
#define KSTEP  32
#define KSTEPS (KDIM / KSTEP)
#define HPITCH 2568          // padded fp16 strip pitch (halfs)
#define TPW    10            // column tiles per wave (16 waves * 10 * 16 = 2560)
#define THREADS 512
#define WT_BYTES ((size_t)NCOL * KDIM * 2)   // bf16 transposed weights
#define TILE_STRIDE (16 * (KDIM / 2) * 4)    // 49152 B between column tiles in wT

__device__ __forceinline__ unsigned short f32_to_bf16_rne(float f) {
  union { float f; unsigned int u; } v; v.f = f;
  unsigned int u = v.u;
  return (unsigned short)((u + 0x7FFFu + ((u >> 16) & 1u)) >> 16);
}
__device__ __forceinline__ unsigned int pack_bf16(float lo, float hi) {
  return ((unsigned int)f32_to_bf16_rne(hi) << 16) | (unsigned int)f32_to_bf16_rne(lo);
}
__device__ __forceinline__ float sigmoid_fast(float x) {
  return 1.0f / (1.0f + __expf(-x));
}
__device__ __forceinline__ float tanh_fast(float x) {
  x = fminf(fmaxf(x, -15.0f), 15.0f);
  float e = __expf(2.0f * x);
  return (e - 1.0f) / (e + 1.0f);
}

// Kernel 0a: w (KDIM x NCOL f32) -> wT (NCOL x KDIM bf16): K contiguous per column.
__global__ void wt_convert_kernel(const float* __restrict__ w,
                                  unsigned short* __restrict__ wt) {
  int idx = blockIdx.x * blockDim.x + threadIdx.x;
  if (idx >= NCOL * KDIM) return;
  int n = idx / KDIM;
  int k = idx - n * KDIM;
  wt[idx] = f32_to_bf16_rne(w[(size_t)k * NCOL + n]);
}

// Kernel 0b: pack concat(inputs,h1,h2) -> xb (B x KDIM bf16), one u32 pair/thread.
__global__ void x_convert_kernel(const float* __restrict__ inputs,
                                 const float* __restrict__ h1,
                                 const float* __restrict__ h2,
                                 unsigned int* __restrict__ xb_pairs, int nrows) {
  int idx = blockIdx.x * blockDim.x + threadIdx.x;
  if (idx >= nrows * (KDIM / 2)) return;
  int row = idx / (KDIM / 2);
  int k   = (idx - row * (KDIM / 2)) * 2;
  const float* src;
  if (k < UDIM)            src = inputs + (size_t)row * UDIM + k;
  else if (k < 2 * UDIM)   src = h1 + (size_t)row * UDIM + (k - UDIM);
  else                     src = h2 + (size_t)row * UDIM + (k - 2 * UDIM);
  float2 f = *(const float2*)src;
  xb_pairs[idx] = pack_bf16(f.x, f.y);
}

// Kernel 1: 32-row strip per workgroup; fused GEMM + LN + gates.
// GVS addressing (uniform SGPR base + 32-bit lane offset + folded ioffset),
// explicit double-buffered B so WMMA runs one tile behind its loads.
__global__ __launch_bounds__(THREADS, 1)
void lstm_fused_kernel(const unsigned short* __restrict__ xb,  // bf16 B x KDIM
                       const float* __restrict__ c1,
                       const float* __restrict__ c2,
                       const unsigned short* __restrict__ wt,  // bf16 NCOL x KDIM
                       const float* __restrict__ bias,         // 1 x NCOL
                       const float* __restrict__ norm_w,       // 5 x U
                       const float* __restrict__ norm_b,       // 5 x U
                       float* __restrict__ out_c,
                       float* __restrict__ out_h) {
  extern __shared__ char smem[];
  _Float16* strip  = (_Float16*)smem;                          // MTILE*HPITCH f16
  float*    s_mean = (float*)(smem + MTILE * HPITCH * 2);
  float*    s_rstd = s_mean + MTILE * 5;

  const int tid  = threadIdx.x;
  const int wave = tid >> 5;
  const int lane = tid & 31;
  const int row0 = blockIdx.x * MTILE;

  const int half   = lane >> 4;      // K-half selector (this code's contiguous k = 16*half + s, A and B)
  const int lane16 = lane & 15;      // A: row index, B/C: column index

  v8f acc[RT][TPW];
  #pragma unroll
  for (int r = 0; r < RT; ++r)
    #pragma unroll
    for (int t = 0; t < TPW; ++t) { v8f z = {}; acc[r][t] = z; }

  const char* abase = (const char*)xb;     // uniform, advances 64 B per K-step
  const char* bbase = (const char*)wt;     // uniform, advances 64 B per K-step
  unsigned int aoff[RT];
  #pragma unroll
  for (int r = 0; r < RT; ++r)
    aoff[r] = (unsigned int)((row0 + r * 16 + lane16) * (KDIM * 2) + half * 32);
  const unsigned int boff =
      (unsigned int)((wave * (TPW * 16) + lane16) * (KDIM * 2) + half * 32);

  for (int ks = 0; ks < KSTEPS; ++ks) {
    // ---- A fragments: two b128 bf16 loads per row-tile ----
    v16bf afrag[RT];
    #pragma unroll
    for (int r = 0; r < RT; ++r) {
      uint4 alo = *(const uint4*)(abase + aoff[r]);
      uint4 ahi = *(const uint4*)(abase + aoff[r] + 16);
      __builtin_prefetch(abase + aoff[r] + 64, 0, 1);   // next K-slab
      v8u au;
      au[0] = alo.x; au[1] = alo.y; au[2] = alo.z; au[3] = alo.w;
      au[4] = ahi.x; au[5] = ahi.y; au[6] = ahi.z; au[7] = ahi.w;
      afrag[r] = __builtin_bit_cast(v16bf, au);
    }

    // ---- B: double-buffered; each fragment feeds RT WMMAs ----
    uint4 Blo[2], Bhi[2];
    Blo[0] = *(const uint4*)(bbase + boff);
    Bhi[0] = *(const uint4*)(bbase + boff + 16);
    #pragma unroll
    for (int t = 0; t < TPW; ++t) {
      const int cur = t & 1;
      if (t + 1 < TPW) {
        Blo[cur ^ 1] = *(const uint4*)(bbase + boff + (t + 1) * TILE_STRIDE);
        Bhi[cur ^ 1] = *(const uint4*)(bbase + boff + (t + 1) * TILE_STRIDE + 16);
      }
      __builtin_prefetch(bbase + boff + t * TILE_STRIDE + 64, 0, 1);
      v8u bu;
      bu[0] = Blo[cur].x; bu[1] = Blo[cur].y; bu[2] = Blo[cur].z; bu[3] = Blo[cur].w;
      bu[4] = Bhi[cur].x; bu[5] = Bhi[cur].y; bu[6] = Bhi[cur].z; bu[7] = Bhi[cur].w;
      v16bf bfrag = __builtin_bit_cast(v16bf, bu);
      #pragma unroll
      for (int r = 0; r < RT; ++r)
        acc[r][t] = __builtin_amdgcn_wmma_f32_16x16x32_bf16(
            false, afrag[r], false, bfrag, (short)0, acc[r][t], false, false);
    }
    abase += KSTEP * 2;   // k0 advance (bytes)
    bbase += KSTEP * 2;
  }

  // ---- accumulators (+bias) -> fp16 LDS strip ----
  #pragma unroll
  for (int t = 0; t < TPW; ++t) {
    const int n = (wave * TPW + t) * 16 + lane16;
    const float bv = bias[n];
    #pragma unroll
    for (int r = 0; r < RT; ++r)
      #pragma unroll
      for (int v = 0; v < 8; ++v)
        strip[(r * 16 + v + half * 8) * HPITCH + n] = (_Float16)(acc[r][t][v] + bv);
  }
  __syncthreads();

  // ---- per-(row,gate) mean / rstd: wave w owns rows 2w, 2w+1 ----
  #pragma unroll
  for (int r = 0; r < RT; ++r) {
    const int rr = wave * RT + r;
    #pragma unroll
    for (int g = 0; g < 5; ++g) {
      float s = 0.0f, s2 = 0.0f;
      #pragma unroll
      for (int i = 0; i < UDIM / 32; ++i) {
        float x = (float)strip[rr * HPITCH + g * UDIM + lane + 32 * i];
        s += x; s2 += x * x;
      }
      #pragma unroll
      for (int m = 16; m >= 1; m >>= 1) {
        s  += __shfl_xor(s,  m, 32);
        s2 += __shfl_xor(s2, m, 32);
      }
      if (lane == 0) {
        float mean = s * (1.0f / UDIM);
        float var  = s2 * (1.0f / UDIM) - mean * mean;
        s_mean[rr * 5 + g] = mean;
        s_rstd[rr * 5 + g] = rsqrtf(var + 1e-5f);
      }
    }
  }
  __syncthreads();

  // ---- fused LN + gates + cell update, stream outputs ----
  #pragma unroll 4
  for (int it = 0; it < (MTILE * UDIM) / THREADS; ++it) {
    const int idx = it * THREADS + tid;
    const int r = idx >> 9;            // / UDIM
    const int u = idx & (UDIM - 1);
    float ln[5];
    #pragma unroll
    for (int g = 0; g < 5; ++g) {
      float x = (float)strip[r * HPITCH + g * UDIM + u];
      ln[g] = (x - s_mean[r * 5 + g]) * s_rstd[r * 5 + g] * norm_w[g * UDIM + u]
              + norm_b[g * UDIM + u];
    }
    const size_t gi = (size_t)(row0 + r) * UDIM + u;
    float nc = c1[gi] * sigmoid_fast(ln[2]) + c2[gi] * sigmoid_fast(ln[3])
             + sigmoid_fast(ln[0]) * tanh_fast(ln[1]);
    float nh = nc * sigmoid_fast(ln[4]);
    out_c[gi] = nc;
    out_h[gi] = nh;
  }
}

extern "C" void kernel_launch(void* const* d_in, const int* in_sizes, int n_in,
                              void* d_out, int out_size, void* d_ws, size_t ws_size,
                              hipStream_t stream) {
  (void)n_in; (void)out_size; (void)ws_size;
  const float* inputs = (const float*)d_in[0];
  const float* c1     = (const float*)d_in[1];
  const float* h1     = (const float*)d_in[2];
  const float* c2     = (const float*)d_in[3];
  const float* h2     = (const float*)d_in[4];
  const float* w      = (const float*)d_in[5];
  const float* b      = (const float*)d_in[6];
  const float* nw     = (const float*)d_in[7];
  const float* nb     = (const float*)d_in[8];

  const int B = in_sizes[0] / UDIM;            // 16384
  unsigned short* wt = (unsigned short*)d_ws;                      // 7.9 MB
  unsigned short* xb = (unsigned short*)((char*)d_ws + WT_BYTES);  // 50.3 MB
  float* out_c = (float*)d_out;
  float* out_h = out_c + (size_t)B * UDIM;

  const size_t smem_bytes = (size_t)MTILE * HPITCH * 2 + 2 * MTILE * 5 * 4;
  (void)hipFuncSetAttribute((const void*)lstm_fused_kernel,
                            hipFuncAttributeMaxDynamicSharedMemorySize,
                            (int)smem_bytes);

  wt_convert_kernel<<<(NCOL * KDIM + 255) / 256, 256, 0, stream>>>(w, wt);
  x_convert_kernel<<<(B * (KDIM / 2) + 255) / 256, 256, 0, stream>>>(
      inputs, h1, h2, (unsigned int*)xb, B);
  lstm_fused_kernel<<<B / MTILE, THREADS, smem_bytes, stream>>>(
      xb, c1, c2, wt, b, nw, nb, out_c, out_h);
}
